// MotionEstimation_23854248362627
// MI455X (gfx1250) — matrix operand order repaired and checked by reading, not compiled, not run on hardware
//
#include <hip/hip_runtime.h>
#include <math.h>

// ---------------------------------------------------------------------------
// Deformable-conv U-Net for MI455X (gfx1250, wave32).
// Convs: implicit GEMM on v_wmma_f32_16x16x32_f16 (f32 accumulate);
// weights pre-packed in exact per-lane WMMA fragment order (f16), staged to
// LDS with gfx1250 async DMA; input halo slabs staged to LDS as f16 with
// incremental (division-free) fragment addressing; padded channel strides
// make epilogue stores unconditional.
// ---------------------------------------------------------------------------

typedef __attribute__((ext_vector_type(16))) _Float16 v16h;
typedef __attribute__((ext_vector_type(8)))  _Float16 v8h;
typedef __attribute__((ext_vector_type(8)))  float    v8f;

union V16H { v16h v; _Float16 e[16]; };
union V8F  { v8f  v; float    e[8]; };

#ifdef __has_builtin
#if __has_builtin(__builtin_amdgcn_global_load_async_to_lds_b128) && \
    __has_builtin(__builtin_amdgcn_s_wait_asynccnt)
#define USE_ASYNC_LDS 1
#endif
#endif

#ifdef USE_ASYNC_LDS
typedef int v4i_vs __attribute__((vector_size(16)));
typedef __attribute__((address_space(1))) v4i_vs* gas1_v4i;
typedef __attribute__((address_space(3))) v4i_vs* las3_v4i;
#endif

__device__ __forceinline__ float fast_silu(float v) {
  return v * __builtin_amdgcn_rcpf(1.0f + __expf(-v));   // x * sigmoid(x)
}

// ---------------------------------------------------------------------------
// Weight pre-pack: OIHW fp32 -> f16 in exact per-lane A-fragment order:
//   [mt][kc][mr(16)][lh(2)][h(16)],  K(h) = kc*32 + lh*16 + h
// so the conv kernel loads one contiguous, 32B-aligned v16h per K-chunk.
// ---------------------------------------------------------------------------
__global__ __launch_bounds__(256)
void pack_weights_kernel(const float* __restrict__ w, _Float16* __restrict__ out,
                         int Cout, int Ktot, int KC, int tot)
{
  int idx = blockIdx.x * 256 + threadIdx.x;
  if (idx >= tot) return;
  const int h    = idx & 15;
  const int lh   = (idx >> 4) & 1;
  const int mr   = (idx >> 5) & 15;
  const int rest = idx >> 9;
  const int kc   = rest % KC;
  const int mt   = rest / KC;
  const int k    = kc * 32 + lh * 16 + h;
  const int m    = mt * 16 + mr;
  const int mc   = min(m, Cout - 1);
  const int kc2  = min(k, Ktot - 1);
  const float v  = w[(size_t)mc * Ktot + kc2];
  out[idx] = (_Float16)(((m < Cout) && (k < Ktot)) ? v : 0.0f);
}

// ---------------------------------------------------------------------------
// Implicit-GEMM conv2d, stride 1, pad=(KSZ-1)/2, NCHW with padded C-strides.
// Block = 8 waves over 128 consecutive outputs of one row segment.
// ---------------------------------------------------------------------------
template<int KSZ, int ACT, int GUARD>
__global__ __launch_bounds__(256)
void conv2d_wmma_kernel(const float* __restrict__ in,
                        const _Float16* __restrict__ wpack,
                        const float* __restrict__ bias, float* __restrict__ out,
                        int Cin, int Cins, int H, int W, int Wsh,
                        int Cout, int Couts, int Ktotp)
{
  constexpr int PAD = (KSZ - 1) / 2;
  constexpr int KK  = KSZ * KSZ;
  constexpr int SEG = 128;
  constexpr int WS  = SEG + 2 * PAD;

  __shared__ __align__(64) _Float16 lds_w[16 * 480];  // KC*512 halves (<=7680)
  __shared__ __align__(64) _Float16 lds_x[14336];     // Rp*WS (<=12672)

  const int HW    = H * W;
  const int Ktot  = Cin * KK;
  const int p0    = blockIdx.x * SEG;
  const int m0    = blockIdx.y * 16;
  const int b     = blockIdx.z;
  const int tid   = threadIdx.x;
  const int lane  = tid & 31;
  const int wavei = tid >> 5;
  const int lhalf = lane >> 4;
  const int l15   = lane & 15;

  const float* __restrict__ inb = in + (size_t)b * Cins * HW;
  __builtin_prefetch(inb + p0, 0, 3);      // global_prefetch_b8

  // ---- stage pre-packed weight tile (contiguous f16, fragment order) ----
  const _Float16* __restrict__ wsrc = wpack + (size_t)blockIdx.y * 16 * Ktotp;
  const int wbytes = 16 * Ktotp * 2;       // multiple of 1024
#ifdef USE_ASYNC_LDS
  if (wavei == 0) {
    for (int o = lane * 16; o < wbytes; o += 32 * 16) {
      __builtin_amdgcn_global_load_async_to_lds_b128(
          (gas1_v4i)(void*)const_cast<char*>((const char*)wsrc + o),
          (las3_v4i)((char*)lds_w + o),
          0, 0);
    }
    __builtin_amdgcn_s_wait_asynccnt(0);
  }
#else
  for (int i = tid * 8; i < 16 * Ktotp; i += 256 * 8)
    *(v8h*)(lds_w + i) = *(const v8h*)(wsrc + i);
#endif

  // ---- hoisted, clamped bias loads (batched; consumed in epilogue) ----
  float bv[8];
  #pragma unroll
  for (int vI = 0; vI < 8; ++vI)
    bv[vI] = bias[min(m0 + vI + lhalf * 8, Cout - 1)];

  // ---- stage input halo slab as f16: lds_x[row=k/KSZ][x]; zero pad rows ----
  const int oy  = p0 >> Wsh;
  const int ox0 = p0 & (W - 1);
  const int R   = Cin * KSZ;                    // real rows
  const int Rp  = (Ktotp + KSZ - 1) / KSZ;      // incl. zero pad rows
  for (int row = wavei; row < Rp; row += 8) {
    _Float16* __restrict__ dst = lds_x + row * WS;
    if (row < R) {
      const int ci = row / KSZ;                 // compile-time divisor
      const int ky = row - ci * KSZ;
      const int iy = oy + ky - PAD;
      const bool oky = (iy >= 0) && (iy < H);
      const int iyc = min(max(iy, 0), H - 1);
      const float* __restrict__ src = inb + (size_t)ci * HW + (size_t)iyc * W;
      for (int xx = lane; xx < WS; xx += 32) {
        const int ix = ox0 + xx - PAD;
        const bool ok = oky && (ix >= 0) && (ix < W);
        const int ixc = min(max(ix, 0), W - 1);
        const float v = src[ixc];               // unconditional, clamped
        dst[xx] = (_Float16)(ok ? v : 0.0f);
      }
    } else {
      for (int xx = lane; xx < WS; xx += 32) dst[xx] = (_Float16)0.0f;
    }
  }
  __syncthreads();

  // ---- WMMA main loop: one 16x16 tile per wave, K in chunks of 32 ----
  const int xl = wavei * 16 + l15;
  // A fragments: contiguous per-lane v16h, advancing 512 halves per chunk.
  const _Float16* __restrict__ aptr = lds_w + l15 * 32 + lhalf * 16;

  v8f acc = {};
  for (int kb = 0; kb < Ktotp; kb += 32) {
    V16H A, Bm;
    A.v = *(const v16h*)aptr;
    aptr += 512;
    // B fragment: incremental (division-free) addressing over k = kb+h(+16).
    {
      const int k0 = kb + lhalf * 16;
      int row  = k0 / KSZ;                      // const-divisor, once/chunk
      int kx   = k0 - row * KSZ;
      int addr = row * WS + xl + kx;
      #pragma unroll
      for (int h = 0; h < 16; ++h) {
        Bm.e[h] = lds_x[addr];
        kx++;
        const bool wrap = (kx == KSZ);
        addr += wrap ? (WS - KSZ + 1) : 1;
        kx = wrap ? 0 : kx;
      }
    }
    acc = __builtin_amdgcn_wmma_f32_16x16x32_f16(false, A.v, false, Bm.v,
                                                 (short)0, acc, false, false);
  }

  // ---- epilogue: base pointer + HW stride per channel row ----
  V8F d; d.v = acc;
  const int nOut = p0 + xl;
  const int Ceff = GUARD ? Cout : Couts;
  float* __restrict__ op = out + ((size_t)b * Ceff + (m0 + lhalf * 8)) * HW + nOut;
  #pragma unroll
  for (int vI = 0; vI < 8; ++vI) {
    const int m = m0 + vI + lhalf * 8;
    float val = d.e[vI] + bv[vI];
    if (ACT) val = fast_silu(val);
    if (!GUARD || (m < Cout)) *op = val;
    op += HW;
  }
}

// ---------------------------------------------------------------------------
// torchvision-semantics DeformConv2d: 3x3, stride 1, pad 1, no bias.
// ---------------------------------------------------------------------------
__global__ __launch_bounds__(256)
void deform_conv_kernel(const float* __restrict__ x, const float* __restrict__ off,
                        const float* __restrict__ wgt, float* __restrict__ out,
                        int Cin, int xs, int offs, int H, int W, int Wsh,
                        int Cout, int groups, int OG)
{
  const int HW  = H * W;
  const int pos = blockIdx.x * 256 + threadIdx.x;
  if (pos >= HW) return;
  const int co = blockIdx.y;
  const int b  = blockIdx.z;
  const int yq = pos >> Wsh;
  const int xq = pos & (W - 1);

  const int cpg = Cin / groups;
  const int g   = co / (Cout / groups);
  const int Cog = Cin / OG;
  const int og  = (g * cpg) / Cog;

  const float* __restrict__ offb = off + ((size_t)b * offs) * HW + pos;
  const float* __restrict__ xb   = x   + (size_t)b * xs * HW;

  float acc = 0.0f;
  #pragma unroll
  for (int tap = 0; tap < 9; ++tap) {
    const int ki = tap / 3 - 1;
    const int kj = tap % 3 - 1;
    const float dy = offb[(size_t)((og * 9 + tap) * 2 + 0) * HW];
    const float dx = offb[(size_t)((og * 9 + tap) * 2 + 1) * HW];
    const float py = (float)(yq + ki) + dy;
    const float px = (float)(xq + kj) + dx;
    const float y0f = floorf(py), x0f = floorf(px);
    const float ly = py - y0f,   lx = px - x0f;
    const int y0 = (int)y0f, x0 = (int)x0f;
    const int y1 = y0 + 1,   x1 = x0 + 1;
    const bool vy0 = (y0 >= 0) && (y0 < H);
    const bool vy1 = (y1 >= 0) && (y1 < H);
    const bool vx0 = (x0 >= 0) && (x0 < W);
    const bool vx1 = (x1 >= 0) && (x1 < W);
    const int cy0 = min(max(y0, 0), H - 1), cy1 = min(max(y1, 0), H - 1);
    const int cx0 = min(max(x0, 0), W - 1), cx1 = min(max(x1, 0), W - 1);
    const float w00 = (1.0f - ly) * (1.0f - lx) * ((vy0 && vx0) ? 1.0f : 0.0f);
    const float w01 = (1.0f - ly) * lx          * ((vy0 && vx1) ? 1.0f : 0.0f);
    const float w10 = ly * (1.0f - lx)          * ((vy1 && vx0) ? 1.0f : 0.0f);
    const float w11 = ly * lx                   * ((vy1 && vx1) ? 1.0f : 0.0f);
    const int i00 = cy0 * W + cx0, i01 = cy0 * W + cx1;
    const int i10 = cy1 * W + cx0, i11 = cy1 * W + cx1;
    for (int cl = 0; cl < cpg; ++cl) {
      const int ci = g * cpg + cl;
      const float* __restrict__ p = xb + (size_t)ci * HW;
      const float v = w00 * p[i00] + w01 * p[i01] + w10 * p[i10] + w11 * p[i11];
      acc += wgt[((size_t)co * cpg + cl) * 9 + tap] * v;
    }
  }
  out[((size_t)b * Cout + co) * HW + pos] = acc;
}

// ---------------------------------------------------------------------------
__global__ __launch_bounds__(256)
void avgpool2_kernel(const float* __restrict__ in, float* __restrict__ out,
                     int C, int Cins, int H, int W, int Wosh)
{
  const int Ho = H >> 1, Wo = W >> 1;
  const int pos = blockIdx.x * 256 + threadIdx.x;
  if (pos >= Ho * Wo) return;
  const int c = blockIdx.y;
  const int b = blockIdx.z;
  const int yo = pos >> Wosh;
  const int xo = pos & (Wo - 1);
  const float* __restrict__ p =
      in + (((size_t)b * Cins + c) * H + (size_t)yo * 2) * W + (size_t)xo * 2;
  out[((size_t)b * C + c) * (Ho * Wo) + pos] = 0.25f * (p[0] + p[1] + p[W] + p[W + 1]);
}

__global__ __launch_bounds__(256)
void up2x_kernel(const float* __restrict__ in, float* __restrict__ out,
                 int C, int Cins, int H, int W, int Wosh)
{
  const int Ho = 2 * H, Wo = 2 * W;
  const int pos = blockIdx.x * 256 + threadIdx.x;
  if (pos >= Ho * Wo) return;
  const int c = blockIdx.y;
  const int b = blockIdx.z;
  const int yo = pos >> Wosh;
  const int xo = pos & (Wo - 1);
  const float sy = (float)yo * (float)(H - 1) / (float)(Ho - 1);
  const float sx = (float)xo * (float)(W - 1) / (float)(Wo - 1);
  const int y0 = (int)floorf(sy); const int y1 = min(y0 + 1, H - 1);
  const int x0 = (int)floorf(sx); const int x1 = min(x0 + 1, W - 1);
  const float wy = sy - (float)y0;
  const float wx = sx - (float)x0;
  const float* __restrict__ p = in + ((size_t)b * Cins + c) * H * W;
  const float r0 = p[(size_t)y0 * W + x0] * (1.0f - wx) + p[(size_t)y0 * W + x1] * wx;
  const float r1 = p[(size_t)y1 * W + x0] * (1.0f - wx) + p[(size_t)y1 * W + x1] * wx;
  out[((size_t)b * C + c) * (Ho * Wo) + pos] = r0 * (1.0f - wy) + r1 * wy;
}

__global__ __launch_bounds__(256)
void copy_channels_kernel(const float* __restrict__ src, float* __restrict__ dst,
                          int Cs, int HW, int Cdst, int c0)
{
  const int pos = blockIdx.x * 256 + threadIdx.x;
  if (pos >= HW) return;
  const int c = blockIdx.y;
  const int b = blockIdx.z;
  dst[((size_t)b * Cdst + (c0 + c)) * HW + pos] = src[((size_t)b * Cs + c) * HW + pos];
}

// ---------------------------------------------------------------------------
// Host-side launch helpers
// ---------------------------------------------------------------------------
static int log2i(int w) { return (w == 512) ? 9 : ((w == 256) ? 8 : 7); }

static void conv(_Float16** cursor, const float* in, const float* w, const float* bias,
                 float* out, int B, int Cin, int Cins, int H, int W, int Cout,
                 int ksz, int act, int guard, hipStream_t s)
{
  const int KK    = ksz * ksz;
  const int Ktot  = Cin * KK;
  const int Ktotp = (Ktot + 31) & ~31;
  const int KC    = Ktotp / 32;
  const int Mt    = (Cout + 15) / 16;
  const int Couts = Mt * 16;
  _Float16* wp = *cursor;
  *cursor += (size_t)Mt * 16 * Ktotp;
  const int tot = Mt * 16 * Ktotp;
  pack_weights_kernel<<<(tot + 255) / 256, 256, 0, s>>>(w, wp, Cout, Ktot, KC, tot);

  dim3 grid(H * W / 128, Mt, B);
  dim3 blk(256);
  const int Wsh = log2i(W);
  if (guard) {
    conv2d_wmma_kernel<3, 1, 1><<<grid, blk, 0, s>>>(in, wp, bias, out, Cin, Cins, H, W, Wsh, Cout, Couts, Ktotp);
  } else if (ksz == 3) {
    if (act) conv2d_wmma_kernel<3, 1, 0><<<grid, blk, 0, s>>>(in, wp, bias, out, Cin, Cins, H, W, Wsh, Cout, Couts, Ktotp);
    else     conv2d_wmma_kernel<3, 0, 0><<<grid, blk, 0, s>>>(in, wp, bias, out, Cin, Cins, H, W, Wsh, Cout, Couts, Ktotp);
  } else {
    if (act) conv2d_wmma_kernel<5, 1, 0><<<grid, blk, 0, s>>>(in, wp, bias, out, Cin, Cins, H, W, Wsh, Cout, Couts, Ktotp);
    else     conv2d_wmma_kernel<5, 0, 0><<<grid, blk, 0, s>>>(in, wp, bias, out, Cin, Cins, H, W, Wsh, Cout, Couts, Ktotp);
  }
}

static void deform(const float* x, const float* off, const float* w, float* out,
                   int B, int Cin, int xs, int offs, int H, int W, int Cout,
                   int groups, hipStream_t s)
{
  dim3 grid((H * W + 255) / 256, Cout, B);
  deform_conv_kernel<<<grid, 256, 0, s>>>(x, off, w, out, Cin, xs, offs, H, W,
                                          log2i(W), Cout, groups, 2);
}

static void avgpool_launch(const float* in, float* out, int B, int C, int Cins,
                           int H, int W, hipStream_t s)
{
  dim3 grid((H / 2) * (W / 2) / 256, C, B);
  avgpool2_kernel<<<grid, 256, 0, s>>>(in, out, C, Cins, H, W, log2i(W) - 1);
}

static void up2_launch(const float* in, float* out, int B, int C, int Cins,
                       int H, int W, hipStream_t s)
{
  dim3 grid(4 * H * W / 256, C, B);
  up2x_kernel<<<grid, 256, 0, s>>>(in, out, C, Cins, H, W, log2i(W) + 1);
}

static void ccopy(const float* src, float* dst, int B, int C, int Cs, int HW,
                  int Cdst, int c0, hipStream_t s)
{
  dim3 grid(HW / 256, C, B);
  copy_channels_kernel<<<grid, 256, 0, s>>>(src, dst, Cs, HW, Cdst, c0);
}

// ---------------------------------------------------------------------------
extern "C" void kernel_launch(void* const* d_in, const int* in_sizes, int n_in,
                              void* d_out, int out_size, void* d_ws, size_t ws_size,
                              hipStream_t stream)
{
  (void)in_sizes; (void)n_in; (void)out_size; (void)ws_size;
  const float* x = (const float*)d_in[0];
#define IN(i) ((const float*)d_in[i])

  const size_t S1 = (size_t)4 * 512 * 512;
  const size_t S2 = (size_t)4 * 256 * 256;
  const size_t S3 = (size_t)4 * 128 * 128;
  float* ws     = (float*)d_ws;
  float* eo_mid = ws;                 // stride 32 @ S1 (reused at all levels)
  float* offb   = eo_mid + 32 * S1;   // stride 48 @ S1 (reused)
  float* comp   = offb   + 48 * S1;   // deform out, unpadded, max 2*S1
  float* mid    = comp   +  2 * S1;   // stride 16, max @ S1
  float* c1d    = mid    + 16 * S1;   // stride 16 @ S1 (skip, persists)
  float* pool1  = c1d    + 16 * S1;   // 8 @ S2 (unpadded)
  float* c2d    = pool1  +  8 * S2;   // stride 16 @ S2 (skip, persists)
  float* pool2b = c2d    + 16 * S2;   // 16 @ S3
  float* c3d    = pool2b + 16 * S3;   // stride 16 @ S3
  float* upb    = c3d    + 16 * S3;   // up2x out, unpadded, max 8*S1
  float* usb    = upb    +  8 * S1;   // stride 16, max @ S1
  float* catb   = usb    + 16 * S1;   // concat, unpadded, max 16*S1
  float* c2u    = catb   + 16 * S1;   // stride 16 @ S2
  _Float16* cur = (_Float16*)(c2u + 16 * S2);   // f16 weight-pack arena

  // ---- level 1 down (512x512) ----
  conv(&cur, x,      IN(1), IN(2), eo_mid, 4,  4,  4, 512, 512, 18, 5, 0, 0, stream);
  conv(&cur, eo_mid, IN(3), IN(4), offb,   4, 18, 32, 512, 512, 36, 5, 0, 0, stream);
  deform(x, offb, IN(5), comp, 4, 4, 4, 48, 512, 512, 2, 2, stream);
  conv(&cur, comp, IN(6), IN(7), mid, 4, 2,  2, 512, 512, 4, 3, 1, 0, stream);
  conv(&cur, mid,  IN(8), IN(9), c1d, 4, 4, 16, 512, 512, 8, 3, 1, 0, stream);
  avgpool_launch(c1d, pool1, 4, 8, 16, 512, 512, stream);

  // ---- level 2 down (256x256) ----
  conv(&cur, pool1,  IN(10), IN(11), eo_mid, 4,  8,  8, 256, 256, 18, 5, 0, 0, stream);
  conv(&cur, eo_mid, IN(12), IN(13), offb,   4, 18, 32, 256, 256, 36, 5, 0, 0, stream);
  deform(pool1, offb, IN(14), comp, 4, 8, 8, 48, 256, 256, 8, 2, stream);
  conv(&cur, comp, IN(15), IN(16), mid, 4,  8,  8, 256, 256, 12, 3, 1, 0, stream);
  conv(&cur, mid,  IN(17), IN(18), c2d, 4, 12, 16, 256, 256, 16, 3, 1, 0, stream);
  avgpool_launch(c2d, pool2b, 4, 16, 16, 256, 256, stream);

  // ---- level 3 bottleneck (128x128) ----
  conv(&cur, pool2b, IN(19), IN(20), eo_mid, 4, 16, 16, 128, 128, 18, 5, 0, 0, stream);
  conv(&cur, eo_mid, IN(21), IN(22), offb,   4, 18, 32, 128, 128, 36, 5, 0, 0, stream);
  deform(pool2b, offb, IN(23), comp, 4, 16, 16, 48, 128, 128, 16, 2, stream);
  conv(&cur, comp, IN(24), IN(25), mid, 4, 16, 16, 128, 128, 14, 3, 1, 0, stream);
  conv(&cur, mid,  IN(26), IN(27), c3d, 4, 14, 16, 128, 128, 12, 3, 1, 0, stream);

  // ---- up path: 128 -> 256 ----
  up2_launch(c3d, upb, 4, 12, 16, 128, 128, stream);
  conv(&cur, upb, IN(28), IN(29), usb, 4, 12, 12, 256, 256, 12, 3, 1, 0, stream);
  ccopy(usb, catb, 4, 12, 16, 256 * 256, 28,  0, stream);
  ccopy(c2d, catb, 4, 16, 16, 256 * 256, 28, 12, stream);
  conv(&cur, catb, IN(30), IN(31), mid, 4, 28, 28, 256, 256, 16, 3, 1, 0, stream);
  conv(&cur, mid,  IN(32), IN(33), c2u, 4, 16, 16, 256, 256,  8, 3, 1, 0, stream);

  // ---- up path: 256 -> 512 ----
  up2_launch(c2u, upb, 4, 8, 16, 256, 256, stream);
  conv(&cur, upb, IN(34), IN(35), usb, 4, 8, 8, 512, 512, 8, 3, 1, 0, stream);
  ccopy(usb, catb, 4, 8, 16, 512 * 512, 16, 0, stream);
  ccopy(c1d, catb, 4, 8, 16, 512 * 512, 16, 8, stream);
  conv(&cur, catb, IN(36), IN(37), mid, 4, 16, 16, 512, 512, 8, 3, 1, 0, stream);
  // final layer: unpadded 2-channel output -> guarded epilogue
  conv(&cur, mid,  IN(38), IN(39), (float*)d_out, 4, 8, 16, 512, 512, 2, 3, 1, 1, stream);
#undef IN
}
